// MambaModel_59931973648885
// MI455X (gfx1250) — matrix operand order repaired and checked by reading, not compiled, not run on hardware
//
#include <hip/hip_runtime.h>
#include <hip/hip_bf16.h>
#include <math.h>

// ---------------------------------------------------------------------------
// Model dims (from reference)
// ---------------------------------------------------------------------------
#define BB   2
#define LL   1024
#define DMv  1024
#define DIv  2048      // EX * DM
#define DSv  16
#define DCv  4
#define DTRv 64        // DM / 16
#define HIDv 2048
#define OUTv 1000
#define NLv  4
#define MTOK (BB * LL) // 2048 tokens

typedef _Float16 v16h __attribute__((ext_vector_type(16)));
typedef _Float16 v8h  __attribute__((ext_vector_type(8)));
typedef float    v8f  __attribute__((ext_vector_type(8)));

__device__ __forceinline__ v8f wmma_f16f32(v16h a, v16h b, v8f c) {
    // D = A(16x32 f16) x B(32x16 f16) + C(16x16 f32)
    return __builtin_amdgcn_wmma_f32_16x16x32_f16(
        /*neg_a=*/false, a, /*neg_b=*/false, b,
        /*c_mod=*/(short)0, c, /*reuse_a=*/false, /*reuse_b=*/false);
}

__device__ __forceinline__ float silu_f(float x) {
    return x / (1.0f + __expf(-x));
}
__device__ __forceinline__ float softplus_f(float x) {
    return (x > 20.0f) ? x : log1pf(__expf(x));
}

// Load a 16x32 f16 A-fragment (per-lane row = lane16):
// lane (m = lane16, half = hlf) holds K = 16h .. 16h+15 (one contiguous run)
__device__ __forceinline__ v16h load_afrag(const _Float16* __restrict__ p, int hlf) {
    v8h lo = *(const v8h*)(p + 16 * hlf);
    v8h hi = *(const v8h*)(p + 16 * hlf + 8);
    return __builtin_shufflevector(lo, hi,
        0,1,2,3,4,5,6,7,8,9,10,11,12,13,14,15);
}
// Load a 32x16 f16 B-fragment from K-major column storage:
// lane (n = lane16, half = hlf) holds K = 16h .. 16h+15 (one contiguous run)
__device__ __forceinline__ v16h load_bfrag(const _Float16* __restrict__ p, int hlf) {
    v8h lo = *(const v8h*)(p + 16 * hlf);
    v8h hi = *(const v8h*)(p + 16 * hlf + 8);
    return __builtin_shufflevector(lo, hi,
        0,1,2,3,4,5,6,7,8,9,10,11,12,13,14,15);
}

// ---------------------------------------------------------------------------
// f32 -> f16 elementwise convert
// ---------------------------------------------------------------------------
__global__ void cvt_f32_to_f16_kernel(const float* __restrict__ in,
                                      _Float16* __restrict__ out, int n) {
    int i = blockIdx.x * 256 + threadIdx.x;
    if (i < n) out[i] = (_Float16)in[i];
}

// ---------------------------------------------------------------------------
// LayerNorm over one row per block (256 threads). Optional f32/f16 outputs,
// optional exact-GELU epilogue (for the hidden head LN).
// ---------------------------------------------------------------------------
__global__ void ln_kernel(const float* __restrict__ in, long in_row_stride,
                          const float* __restrict__ w, const float* __restrict__ bv,
                          float* __restrict__ outF, _Float16* __restrict__ outH,
                          int n, int do_gelu) {
    const int row = blockIdx.x;
    const int tid = threadIdx.x;
    const float* x = in + (long)row * in_row_stride;

    __shared__ float s1[256];
    __shared__ float s2[256];
    float a = 0.f, b = 0.f;
    for (int i = tid; i < n; i += 256) {
        float v = x[i];
        a += v;
        b += v * v;
    }
    s1[tid] = a; s2[tid] = b;
    __syncthreads();
    for (int s = 128; s > 0; s >>= 1) {
        if (tid < s) { s1[tid] += s1[tid + s]; s2[tid] += s2[tid + s]; }
        __syncthreads();
    }
    const float mean = s1[0] / (float)n;
    const float var  = s2[0] / (float)n - mean * mean;
    const float rstd = rsqrtf(var + 1e-5f);

    for (int i = tid; i < n; i += 256) {
        float v = (x[i] - mean) * rstd * w[i] + bv[i];
        if (do_gelu) v = 0.5f * v * (1.0f + erff(v * 0.70710678118654752f));
        if (outF) outF[(long)row * n + i] = v;
        if (outH) outH[(long)row * n + i] = (_Float16)v;
    }
}

// ---------------------------------------------------------------------------
// WMMA GEMM:  C[M,N] = act( A[M,K](f16) * W[N,K](f16)^T + bias + resid )
//   - block: 256 threads = 8 waves; wave w owns rows m0 = (by*8+w)*32
//   - per-wave tile 32(M) x 16*NT(N); NT is a compile-time template param so
//     the K-loop is branch-free (EXEC all-1s, full clause/pipeline scheduling)
//   - NT=4: per K-step 12 b128 loads feed 8 WMMAs (1.5 loads/WMMA)
//   - grid: (Ncovered/(16*NT), M/256); n_base shifts the column window so the
//     N=96 GEMM is covered by one NT=4 launch (cols 0-63) + one NT=2 launch
//     (cols 64-95) with zero runtime guards.
//   - W is row-major (N,K) == B-matrix stored K-major per column, so every
//     fragment load is a contiguous 128-bit global load (no transpose needed).
//   - act: 0 = none, 1 = softplus (for dt_proj)
// ---------------------------------------------------------------------------
template <int NT>
__global__ void gemm_wmma_kernel(const _Float16* __restrict__ A,
                                 const _Float16* __restrict__ Wt,
                                 const float* __restrict__ bias,
                                 const float* __restrict__ resid,
                                 float* __restrict__ C,
                                 int Ndim, int Kdim, int act, int n_base) {
    const int tid    = threadIdx.x;
    const int wave   = tid >> 5;
    const int lane   = tid & 31;
    const int lane16 = lane & 15;
    const int hlf    = lane >> 4;

    const int m0 = (blockIdx.y * 8 + wave) * 32;
    const int n0 = n_base + blockIdx.x * (16 * NT);

    v8f acc[2 * NT];
    #pragma unroll
    for (int i = 0; i < 2 * NT; ++i)
        acc[i] = (v8f){0.f, 0.f, 0.f, 0.f, 0.f, 0.f, 0.f, 0.f};

    const _Float16* arow0 = A + (size_t)(m0 + lane16) * Kdim;
    const _Float16* arow1 = arow0 + (size_t)16 * Kdim;
    const _Float16* brow  = Wt + (size_t)(n0 + lane16) * Kdim;

    for (int kk = 0; kk < Kdim; kk += 32) {
        v16h a0 = load_afrag(arow0 + kk, hlf);
        v16h a1 = load_afrag(arow1 + kk, hlf);
        #pragma unroll
        for (int t = 0; t < NT; ++t) {
            v16h bf = load_bfrag(brow + (size_t)(16 * t) * Kdim + kk, hlf);
            acc[t]      = wmma_f16f32(a0, bf, acc[t]);
            acc[NT + t] = wmma_f16f32(a1, bf, acc[NT + t]);
        }
    }

    // Epilogue: C/D layout — VGPR r, lane: row = mbase + r + 8*hlf, col = n0t + lane16
    #pragma unroll
    for (int t = 0; t < NT; ++t) {
        const int ncol = n0 + 16 * t + lane16;
        const float bval = bias ? bias[ncol] : 0.0f;
        #pragma unroll
        for (int mt = 0; mt < 2; ++mt) {
            const int mbase = m0 + 16 * mt;
            #pragma unroll
            for (int r = 0; r < 8; ++r) {
                const int mrow = mbase + r + 8 * hlf;
                float v = acc[mt * NT + t][r] + bval;
                if (resid) v += resid[(size_t)mrow * Ndim + ncol];
                if (act == 1) v = softplus_f(v);
                C[(size_t)mrow * Ndim + ncol] = v;
            }
        }
    }
}

// ---------------------------------------------------------------------------
// Causal depthwise conv (K=4) + SiLU.  Input: first DI columns of xz (f32,
// row stride 2*DI).  Outputs xs (f32) and xs_h (f16).
// ---------------------------------------------------------------------------
__global__ void conv_silu_kernel(const float* __restrict__ xz,
                                 const float* __restrict__ cw,
                                 const float* __restrict__ cb,
                                 float* __restrict__ xs,
                                 _Float16* __restrict__ xs_h) {
    const int idx = blockIdx.x * 256 + threadIdx.x;   // < B*L*DI = 4194304
    const int d = idx & (DIv - 1);
    const int l = (idx >> 11) & (LL - 1);
    const int b = idx >> 21;
    float s = cb[d];
    #pragma unroll
    for (int k = 0; k < DCv; ++k) {
        int ll = l - (DCv - 1) + k;
        if (ll >= 0)
            s += xz[((size_t)(b * LL + ll) << 12) + d] * cw[d * DCv + k];
    }
    float v = silu_f(s);
    xs[idx]   = v;
    xs_h[idx] = (_Float16)v;
}

// ---------------------------------------------------------------------------
// Extract the dt slice (first 64 cols of xdbl, row stride 96) as f16.
// ---------------------------------------------------------------------------
__global__ void dt_slice_kernel(const float* __restrict__ xdbl,
                                _Float16* __restrict__ dts_h) {
    const int idx = blockIdx.x * 256 + threadIdx.x;   // < MTOK*64
    const int m = idx >> 6;
    const int r = idx & 63;
    dts_h[idx] = (_Float16)xdbl[(size_t)m * 96 + r];
}

// ---------------------------------------------------------------------------
// Selective scan: thread per (b,d) channel, 16-state recurrence in registers,
// B_t/C_t (32 floats/step) broadcast through LDS. Sequential over L.
// ---------------------------------------------------------------------------
__global__ void scan_kernel(const float* __restrict__ xs,
                            const float* __restrict__ dtv,
                            const float* __restrict__ xdbl,
                            const float* __restrict__ A_log,
                            const float* __restrict__ Dvp,
                            float* __restrict__ y) {
    const int d = blockIdx.x * 256 + threadIdx.x;
    const int b = blockIdx.y;

    float Ar[DSv];
    #pragma unroll
    for (int n = 0; n < DSv; ++n) Ar[n] = -__expf(A_log[d * DSv + n]);
    const float Dd = Dvp[d];
    float h[DSv] = {};

    __shared__ float bc[32];

    for (int l = 0; l < LL; ++l) {
        if (threadIdx.x < 32)
            bc[threadIdx.x] = xdbl[(size_t)(b * LL + l) * 96 + 64 + threadIdx.x];
        __syncthreads();
        const size_t row = (size_t)(b * LL + l) * DIv + d;
        const float xv  = xs[row];
        const float dtt = dtv[row];
        float acc = 0.f;
        #pragma unroll
        for (int n = 0; n < DSv; ++n) {
            float dA = __expf(dtt * Ar[n]);
            h[n] = dA * h[n] + dtt * bc[n] * xv;
            acc += h[n] * bc[16 + n];
        }
        y[row] = acc + xv * Dd;
        __syncthreads();
    }
}

// ---------------------------------------------------------------------------
// ym = f16( y * silu(z) ),  z = second DI columns of xz.
// ---------------------------------------------------------------------------
__global__ void gate_kernel(const float* __restrict__ y,
                            const float* __restrict__ xz,
                            _Float16* __restrict__ ym_h) {
    const int idx = blockIdx.x * 256 + threadIdx.x;   // < B*L*DI
    const int d = idx & (DIv - 1);
    const size_t row = (size_t)(idx >> 11);
    const float zv = xz[(row << 12) + DIv + d];
    ym_h[idx] = (_Float16)(y[idx] * silu_f(zv));
}

// ---------------------------------------------------------------------------
// Tiny head FC (M=2): thread per output element.
// ---------------------------------------------------------------------------
__global__ void fc_kernel(const float* __restrict__ in,
                          const float* __restrict__ W,
                          const float* __restrict__ bias,
                          float* __restrict__ out,
                          int rows, int N, int K) {
    const int idx = blockIdx.x * 256 + threadIdx.x;
    if (idx >= rows * N) return;
    const int b = idx / N;
    const int n = idx % N;
    const float* xr = in + (size_t)b * K;
    const float* wr = W + (size_t)n * K;
    float s = bias[n];
    for (int k = 0; k < K; ++k) s += xr[k] * wr[k];
    out[idx] = s;
}

// ---------------------------------------------------------------------------
// Host-side launcher
// ---------------------------------------------------------------------------
extern "C" void kernel_launch(void* const* d_in, const int* in_sizes, int n_in,
                              void* d_out, int out_size, void* d_ws, size_t ws_size,
                              hipStream_t stream) {
    (void)in_sizes; (void)n_in; (void)out_size; (void)ws_size;

    const float* x_in       = (const float*)d_in[0];
    const float* ln_w       = (const float*)d_in[1];
    const float* ln_b       = (const float*)d_in[2];
    const float* in_proj_w  = (const float*)d_in[3];
    const float* conv_w     = (const float*)d_in[4];
    const float* conv_b     = (const float*)d_in[5];
    const float* x_proj_w   = (const float*)d_in[6];
    const float* dt_proj_w  = (const float*)d_in[7];
    const float* dt_proj_b  = (const float*)d_in[8];
    const float* A_log      = (const float*)d_in[9];
    const float* Dv         = (const float*)d_in[10];
    const float* out_proj_w = (const float*)d_in[11];
    const float* fn_w       = (const float*)d_in[12];
    const float* fn_b       = (const float*)d_in[13];
    const float* h1_w       = (const float*)d_in[14];
    const float* h1_b       = (const float*)d_in[15];
    const float* hln_w      = (const float*)d_in[16];
    const float* hln_b      = (const float*)d_in[17];
    const float* h2_w       = (const float*)d_in[18];
    const float* h2_b       = (const float*)d_in[19];

    // ---- workspace layout (all buffers written before read) ----
    size_t off = 0;
    char* base = (char*)d_ws;
    auto take = [&](size_t bytes) -> void* {
        void* p = base + off;
        off += (bytes + 255) & ~(size_t)255;
        return p;
    };
    float*     xA      = (float*)take((size_t)MTOK * DMv * 4);
    float*     xB      = (float*)take((size_t)MTOK * DMv * 4);
    _Float16*  xn_h    = (_Float16*)take((size_t)MTOK * DMv * 2);
    _Float16*  w_in_h  = (_Float16*)take((size_t)2 * DIv * DMv * 2);
    _Float16*  w_xp_h  = (_Float16*)take((size_t)96 * DIv * 2);
    _Float16*  w_dt_h  = (_Float16*)take((size_t)DIv * DTRv * 2);
    _Float16*  w_out_h = (_Float16*)take((size_t)DMv * DIv * 2);
    float*     xz      = (float*)take((size_t)MTOK * 2 * DIv * 4);
    float*     xs      = (float*)take((size_t)MTOK * DIv * 4);
    _Float16*  xs_h    = (_Float16*)take((size_t)MTOK * DIv * 2);
    float*     xdbl    = (float*)take((size_t)MTOK * 96 * 4);
    _Float16*  dts_h   = (_Float16*)take((size_t)MTOK * DTRv * 2);
    float*     dtb     = (float*)take((size_t)MTOK * DIv * 4);
    float*     ybuf    = (float*)take((size_t)MTOK * DIv * 4);
    _Float16*  ym_h    = (_Float16*)take((size_t)MTOK * DIv * 2);
    float*     lastv   = (float*)take((size_t)BB * DMv * 4);
    float*     h1buf   = (float*)take((size_t)BB * HIDv * 4);

    const dim3 blk(256);
    const float* cur = x_in;
    float* bufs[2] = {xA, xB};

    for (int i = 0; i < NLv; ++i) {
        float* nxt = bufs[i & 1];

        // weight conversions to f16 (L2-resident, reused across all M tiles)
        cvt_f32_to_f16_kernel<<<(2 * DIv * DMv) / 256, blk, 0, stream>>>(
            in_proj_w + (size_t)i * 2 * DIv * DMv, w_in_h, 2 * DIv * DMv);
        cvt_f32_to_f16_kernel<<<(96 * DIv) / 256, blk, 0, stream>>>(
            x_proj_w + (size_t)i * 96 * DIv, w_xp_h, 96 * DIv);
        cvt_f32_to_f16_kernel<<<(DIv * DTRv) / 256, blk, 0, stream>>>(
            dt_proj_w + (size_t)i * DIv * DTRv, w_dt_h, DIv * DTRv);
        cvt_f32_to_f16_kernel<<<(DMv * DIv) / 256, blk, 0, stream>>>(
            out_proj_w + (size_t)i * DMv * DIv, w_out_h, DMv * DIv);

        // LayerNorm -> f16 activations
        ln_kernel<<<MTOK, blk, 0, stream>>>(cur, (long)DMv,
            ln_w + (size_t)i * DMv, ln_b + (size_t)i * DMv,
            nullptr, xn_h, DMv, 0);

        // xz = xn @ in_proj_w^T   (2048 x 4096, K=1024)
        gemm_wmma_kernel<4><<<dim3(4096 / 64, MTOK / 256), blk, 0, stream>>>(
            xn_h, w_in_h, nullptr, nullptr, xz, 4096, DMv, 0, 0);

        // depthwise causal conv + SiLU
        conv_silu_kernel<<<(MTOK * DIv) / 256, blk, 0, stream>>>(
            xz, conv_w + (size_t)i * DIv * DCv, conv_b + (size_t)i * DIv,
            xs, xs_h);

        // xdbl = xs @ x_proj_w^T  (2048 x 96, K=2048)
        // cols 0-63 with NT=4, cols 64-95 with NT=2 (no runtime guards)
        gemm_wmma_kernel<4><<<dim3(1, MTOK / 256), blk, 0, stream>>>(
            xs_h, w_xp_h, nullptr, nullptr, xdbl, 96, DIv, 0, 0);
        gemm_wmma_kernel<2><<<dim3(1, MTOK / 256), blk, 0, stream>>>(
            xs_h, w_xp_h, nullptr, nullptr, xdbl, 96, DIv, 0, 64);

        // dt slice -> f16
        dt_slice_kernel<<<(MTOK * DTRv) / 256, blk, 0, stream>>>(xdbl, dts_h);

        // dt = softplus(dt_small @ dt_proj_w^T + b)  (2048 x 2048, K=64)
        gemm_wmma_kernel<4><<<dim3(DIv / 64, MTOK / 256), blk, 0, stream>>>(
            dts_h, w_dt_h, dt_proj_b + (size_t)i * DIv, nullptr, dtb,
            DIv, DTRv, 1, 0);

        // selective scan
        scan_kernel<<<dim3(DIv / 256, BB), blk, 0, stream>>>(
            xs, dtb, xdbl, A_log + (size_t)i * DIv * DSv,
            Dv + (size_t)i * DIv, ybuf);

        // gate: ym = f16(y * silu(z))
        gate_kernel<<<(MTOK * DIv) / 256, blk, 0, stream>>>(ybuf, xz, ym_h);

        // x_next = ym @ out_proj_w^T + residual  (2048 x 1024, K=2048)
        gemm_wmma_kernel<4><<<dim3(DMv / 64, MTOK / 256), blk, 0, stream>>>(
            ym_h, w_out_h, nullptr, cur, nxt, DMv, DIv, 0, 0);

        cur = nxt;
    }

    // final LN applied only to the last token of each batch
    ln_kernel<<<BB, blk, 0, stream>>>(cur + (size_t)(LL - 1) * DMv,
        (long)LL * DMv, fn_w, fn_b, lastv, nullptr, DMv, 0);

    // head: h1 = last @ h1_w^T + b ; LN + exact GELU ; out = h @ h2_w^T + b
    fc_kernel<<<(BB * HIDv + 255) / 256, blk, 0, stream>>>(
        lastv, h1_w, h1_b, h1buf, BB, HIDv, DMv);
    ln_kernel<<<BB, blk, 0, stream>>>(h1buf, (long)HIDv, hln_w, hln_b,
        h1buf, nullptr, HIDv, 1);
    fc_kernel<<<(BB * OUTv + 255) / 256, blk, 0, stream>>>(
        h1buf, h2_w, h2_b, (float*)d_out, BB, OUTv, HIDv);
}
